// GCNSag_2611340116531
// MI455X (gfx1250) — compile-verified, matrix operand order checked by reading, not run on hardware
//
#include <hip/hip_runtime.h>
#include <hip/hip_bf16.h>

// GCN forward on gfx1250:
//   h1 = relu(GCNConv(x, W1, b1)); h2 = relu(GCNConv(h1, W2, b2))
//   hG = segment_mean(concat(h2,h1), batch); out = [hG, log_softmax(hG@Wl+bl)]
// GEMMs use v_wmma_f32_16x16x32_bf16 (confirmed builtin); message passing is
// L2-resident f32 atomics; pooling exploits sorted batch (run-length flush).

typedef __attribute__((ext_vector_type(16))) __bf16 v16bf;
typedef __attribute__((ext_vector_type(8)))  float  v8f;

#define NN 50000
#define EE 800000
#define HH 128
#define NH (NN * HH)

// ---------------- small utility kernels ----------------

__global__ void k_fill(float* __restrict__ p, float v, int n) {
    int i = blockIdx.x * blockDim.x + threadIdx.x;
    if (i < n) p[i] = v;
}

__global__ void k_deg_add(const int* __restrict__ dst, float* __restrict__ deg, int nE) {
    int e = blockIdx.x * blockDim.x + threadIdx.x;
    if (e < nE) atomicAdd(&deg[dst[e]], 1.0f);
}

__global__ void k_rsqrt_ip(float* __restrict__ p, int n) {
    int i = blockIdx.x * blockDim.x + threadIdx.x;
    if (i < n) p[i] = rsqrtf(p[i]);
}

// y[n,f] *= dinv[n]
__global__ void k_scale(float* __restrict__ y, const float* __restrict__ dinv, int n) {
    int i = blockIdx.x * blockDim.x + threadIdx.x;
    if (i < n) y[i] *= dinv[i >> 7];
}

// h[n,f] = relu(dinv[n]*(s[n,f] + y[n,f]) + b[f])
__global__ void k_combine_relu(const float* __restrict__ s, const float* __restrict__ y,
                               const float* __restrict__ dinv, const float* __restrict__ b,
                               float* __restrict__ h, int n) {
    int i = blockIdx.x * blockDim.x + threadIdx.x;
    if (i >= n) return;
    int node = i >> 7, f = i & 127;
    float v = dinv[node] * (s[i] + y[i]) + b[f];
    h[i] = v > 0.f ? v : 0.f;
}

// ---------------- WMMA bf16 GEMM: C[N,128] = A[N,128] @ W[128,128] ----------------
// W is pre-packed into per-lane B-fragment layout: one 32B contiguous load per
// (nt,kk) fragment per lane.

__device__ __forceinline__ int wmma_k(int v, int half) {
    // 16-bit A/B fragment K index for VGPR-pair v (0..7), wave-half (0/1)
    return ((v & 4) ? 16 : 0) + half * 8 + (v & 3) * 2;
}

__global__ void k_pack_w(const float* __restrict__ W, __bf16* __restrict__ Wp) {
    int t = blockIdx.x * blockDim.x + threadIdx.x;      // 0..8191
    if (t >= 8 * 4 * 32 * 8) return;
    int v = t & 7, lane = (t >> 3) & 31, kk = (t >> 8) & 3, nt = t >> 10;
    int half = lane >> 4;
    int col = nt * 16 + (lane & 15);
    int k = kk * 32 + wmma_k(v, half);
    int base = ((nt * 4 + kk) * 32 + lane) * 16 + 2 * v;
    Wp[base]     = (__bf16)W[k * HH + col];
    Wp[base + 1] = (__bf16)W[(k + 1) * HH + col];
}

__global__ void k_gemm_bf16(const float* __restrict__ A, const v16bf* __restrict__ Wp,
                            float* __restrict__ C, int nrows) {
    int wave = threadIdx.x >> 5;
    int lane = threadIdx.x & 31;
    int tile = blockIdx.x * (blockDim.x >> 5) + wave;
    if (tile * 16 >= nrows) return;        // wave-uniform -> EXEC all-ones at WMMA
    int half = lane >> 4;
    int r = tile * 16 + (lane & 15);

    v16bf a[4];
#pragma unroll
    for (int kk = 0; kk < 4; ++kk) {
#pragma unroll
        for (int v = 0; v < 8; ++v) {
            int k = kk * 32 + wmma_k(v, half);
            float2 f = *reinterpret_cast<const float2*>(A + (size_t)r * HH + k);
            a[kk][2 * v]     = (__bf16)f.x;
            a[kk][2 * v + 1] = (__bf16)f.y;
        }
    }

    v8f acc[8] = {};
#pragma unroll
    for (int nt = 0; nt < 8; ++nt) {
#pragma unroll
        for (int kk = 0; kk < 4; ++kk) {
            v16bf b = Wp[(nt * 4 + kk) * 32 + lane];
            acc[nt] = __builtin_amdgcn_wmma_f32_16x16x32_bf16(
                false, a[kk], false, b, (short)0, acc[nt], false, false);
        }
    }

    int rowbase = tile * 16 + half * 8;
#pragma unroll
    for (int nt = 0; nt < 8; ++nt)
#pragma unroll
        for (int j = 0; j < 8; ++j)
            C[(size_t)(rowbase + j) * HH + nt * 16 + (lane & 15)] = acc[nt][j];
}

// ---------------- edge scatter: s[dst] += y[src] ----------------
// one wave per edge; lane handles 4 consecutive features (float4 gather,
// 4x global_atomic_add_f32 that resolve in L2 since |s| = 25.6MB << 192MB)

__global__ void k_scatter(const int* __restrict__ src, const int* __restrict__ dst,
                          const float* __restrict__ y, float* __restrict__ s, int nE) {
    long long gid = (long long)blockIdx.x * blockDim.x + threadIdx.x;
    int e = (int)(gid >> 5);
    if (e >= nE) return;
    int f = ((int)gid & 31) * 4;
    int sn = src[e], dn = dst[e];
    float4 v = *reinterpret_cast<const float4*>(y + (size_t)sn * HH + f);
    float* p = s + (size_t)dn * HH + f;
    atomicAdd(p + 0, v.x);
    atomicAdd(p + 1, v.y);
    atomicAdd(p + 2, v.z);
    atomicAdd(p + 3, v.w);
}

// ---------------- pooling (batch is sorted: run-length flush) ----------------
// computes h2 on the fly from (s2,y2,dinv,b2); thread t owns feature t of
// concat(h2,h1) (t<128 -> h2[t], else h1[t-128]).

#define POOL_NODES 128

__global__ void k_pool(const float* __restrict__ s2, const float* __restrict__ y2,
                       const float* __restrict__ dinv, const float* __restrict__ b2,
                       const float* __restrict__ h1, const int* __restrict__ batch,
                       float* __restrict__ hGsum, float* __restrict__ cnt, int nN) {
    __shared__ int sb[POOL_NODES];
    int n0 = blockIdx.x * POOL_NODES;
    int t = threadIdx.x;
    int nEnd = n0 + POOL_NODES;
    if (nEnd > nN) nEnd = nN;
    for (int i = n0 + t; i < nEnd; i += 256) sb[i - n0] = batch[i];
    __syncthreads();

    int f = t;
    float acc = 0.f, cAcc = 0.f;
    int gcur = sb[0];
    for (int n = n0; n < nEnd; ++n) {
        int g = sb[n - n0];
        if (g != gcur) {
            atomicAdd(&hGsum[gcur * 256 + f], acc);
            if (t == 0) atomicAdd(&cnt[gcur], cAcc);
            acc = 0.f; cAcc = 0.f; gcur = g;
        }
        float v;
        if (f < 128) {
            size_t idx = (size_t)n * HH + f;
            float x = dinv[n] * (s2[idx] + y2[idx]) + b2[f];
            v = x > 0.f ? x : 0.f;
        } else {
            v = h1[(size_t)n * HH + (f - 128)];
        }
        acc += v;
        cAcc += 1.f;
    }
    atomicAdd(&hGsum[gcur * 256 + f], acc);
    if (t == 0) atomicAdd(&cnt[gcur], cAcc);
}

// ---------------- final: mean, linear head, log_softmax ----------------

__global__ void k_finalize(const float* __restrict__ hGsum, const float* __restrict__ cnt,
                           const float* __restrict__ Wl, const float* __restrict__ bl,
                           float* __restrict__ out) {
    __shared__ float hG[2048];
    __shared__ float lg[64];
    int t = threadIdx.x;
    for (int i = t; i < 2048; i += 256) {
        int g = i >> 8;
        float c = cnt[g];
        c = c > 1.f ? c : 1.f;
        float v = hGsum[i] / c;
        hG[i] = v;
        out[i] = v;
    }
    __syncthreads();
    if (t < 40) {
        int g = t / 5, c = t % 5;
        float s = bl[c];
        for (int f = 0; f < 256; ++f) s += hG[g * 256 + f] * Wl[f * 5 + c];
        lg[g * 5 + c] = s;
    }
    __syncthreads();
    if (t < 8) {
        int g = t;
        float m = lg[g * 5];
        for (int c = 1; c < 5; ++c) m = fmaxf(m, lg[g * 5 + c]);
        float sum = 0.f;
        for (int c = 0; c < 5; ++c) sum += expf(lg[g * 5 + c] - m);
        float lse = m + logf(sum);
        for (int c = 0; c < 5; ++c) out[2048 + g * 5 + c] = lg[g * 5 + c] - lse;
    }
}

// ---------------- host launcher ----------------

extern "C" void kernel_launch(void* const* d_in, const int* in_sizes, int n_in,
                              void* d_out, int out_size, void* d_ws, size_t ws_size,
                              hipStream_t stream) {
    const float* x   = (const float*)d_in[0];
    const int*   ei  = (const int*)d_in[1];
    const int* batch = (const int*)d_in[2];
    const float* W1  = (const float*)d_in[3];
    const float* b1  = (const float*)d_in[4];
    const float* W2  = (const float*)d_in[5];
    const float* b2  = (const float*)d_in[6];
    const float* Wl  = (const float*)d_in[7];
    const float* bl  = (const float*)d_in[8];
    float* out = (float*)d_out;

    const int* src = ei;
    const int* dst = ei + EE;

    float* ws = (float*)d_ws;
    size_t off = 0;
    float*  dinv  = ws + off; off += 50048;          // deg -> dinv (in place)
    float*  y     = ws + off; off += (size_t)NH;     // xW * dinv (per layer)
    float*  s     = ws + off; off += (size_t)NH;     // scatter accumulator
    float*  h1    = ws + off; off += (size_t)NH;     // layer-1 activations
    float*  hGsum = ws + off; off += 2048;           // pooled sums [G,2H]
    float*  cnt   = ws + off; off += 64;             // per-graph node counts
    __bf16* Wp    = (__bf16*)(ws + off);             // packed W (16K bf16)

    const int B = 256;
    const int gN   = (NN + B - 1) / B;               // 196
    const int gNH  = NH / B;                         // 25000
    const int gE   = EE / B;                         // 3125
    const int gSc  = (EE * 32) / B;                  // 100000
    const int gGemm = (3125 + 3) / 4;                // 782 blocks x 4 waves
    const int gPool = (NN + POOL_NODES - 1) / POOL_NODES;  // 391

    // init: deg=1 (self loops), s=0, hGsum/cnt=0
    k_fill<<<gN, B, 0, stream>>>(dinv, 1.0f, NN);
    k_fill<<<gNH, B, 0, stream>>>(s, 0.0f, NH);
    k_fill<<<(2112 + B - 1) / B, B, 0, stream>>>(hGsum, 0.0f, 2112);

    // degrees -> dinv
    k_deg_add<<<gE, B, 0, stream>>>(dst, dinv, EE);
    k_rsqrt_ip<<<gN, B, 0, stream>>>(dinv, NN);

    // ---- layer 1 ----
    k_pack_w<<<32, B, 0, stream>>>(W1, Wp);
    k_gemm_bf16<<<gGemm, 128, 0, stream>>>(x, (const v16bf*)Wp, y, NN);
    k_scale<<<gNH, B, 0, stream>>>(y, dinv, NH);
    k_scatter<<<gSc, B, 0, stream>>>(src, dst, y, s, EE);
    k_combine_relu<<<gNH, B, 0, stream>>>(s, y, dinv, b1, h1, NH);

    // ---- layer 2 ----
    k_fill<<<gNH, B, 0, stream>>>(s, 0.0f, NH);
    k_pack_w<<<32, B, 0, stream>>>(W2, Wp);
    k_gemm_bf16<<<gGemm, 128, 0, stream>>>(h1, (const v16bf*)Wp, y, NN);
    k_scale<<<gNH, B, 0, stream>>>(y, dinv, NH);
    k_scatter<<<gSc, B, 0, stream>>>(src, dst, y, s, EE);

    // ---- pooling (h2 computed on the fly) + head ----
    k_pool<<<gPool, B, 0, stream>>>(s, y, dinv, b2, h1, batch, hGsum, cnt, NN);
    k_finalize<<<1, B, 0, stream>>>(hGsum, cnt, Wl, bl, out);
}
